// GAT_Layer_66640712564807
// MI455X (gfx1250) — matrix operand order repaired and checked by reading, not compiled, not run on hardware
//
#include <hip/hip_runtime.h>
#include <cstdint>

typedef __attribute__((ext_vector_type(16))) _Float16 v16h;
typedef __attribute__((ext_vector_type(8)))  float    v8f;

#define N_NODES 4096
#define F_DIM   128
#define HEADS   8
#define U_DIM   8
#define KCOLS   80      // 64 data + 8 sum-exp + 1 deg + 7 pad
#define JSPLIT  4

union AFrag { v16h v; _Float16 h[16]; };
union BFrag { v16h v; uint4 q[2]; };

// ---------------------------------------------------------------------------
// Stage 1: HW = H@W per head, e_l/e_r, build Mt[80][4096] (f16, transposed):
//   Mt[h*8+u][j] = exp(e_r[h,j]) * HW[h,j,u]
//   Mt[64+h][j]  = exp(e_r[h,j])
//   Mt[72][j]    = 1,  Mt[73..79][j] = 0 (pad)
// Also store e_l[h][j] (f32) for the epilogue.
// ---------------------------------------------------------------------------
__global__ __launch_bounds__(256)
void gat_stage1(const float* __restrict__ H, const float* __restrict__ W,
                const float* __restrict__ aL, const float* __restrict__ aR,
                _Float16* __restrict__ Mt, float* __restrict__ el_ws) {
    __shared__ float s_hw[256];
    const int node = blockIdx.x * 4 + (threadIdx.x >> 6);
    const int t64  = threadIdx.x & 63;
    const int h = t64 >> 3, u = t64 & 7;

    const float* hrow = H + (size_t)node * F_DIM;
    const float* wcol = W + (size_t)h * F_DIM * U_DIM + u;
    float hw = 0.f;
#pragma unroll 8
    for (int f = 0; f < F_DIM; ++f)
        hw += hrow[f] * wcol[(size_t)f * U_DIM];

    s_hw[threadIdx.x] = hw;
    __syncthreads();

    const int base = threadIdx.x & ~7;   // 8 u-lanes of this (node, h)
    float el = 0.f, er = 0.f;
#pragma unroll
    for (int k = 0; k < 8; ++k) {
        float v = s_hw[base + k];
        el += v * aL[h * 8 + k];
        er += v * aR[h * 8 + k];
    }
    float eer = expf(er);

    Mt[(size_t)(h * 8 + u) * N_NODES + node] = (_Float16)(eer * hw);
    if (u == 0) {
        Mt[(size_t)(64 + h) * N_NODES + node] = (_Float16)eer;
        el_ws[(size_t)h * N_NODES + node] = el;
    }
    if (t64 == 0) Mt[(size_t)72 * N_NODES + node] = (_Float16)1.0f;
    if (t64 < 7)  Mt[(size_t)(73 + t64) * N_NODES + node] = (_Float16)0.0f;
}

// ---------------------------------------------------------------------------
// Stage 2: Y = A @ M via v_wmma_f32_16x16x32_f16.
// One wave = 16-row strip x 5 column tiles (80 cols), over a 1024-wide j chunk.
// Partial results go to Ypart[jchunk][4096][80] (deterministic, no atomics).
// ---------------------------------------------------------------------------
__global__ __launch_bounds__(256)
void gat_stage2(const float* __restrict__ A, const _Float16* __restrict__ Mt,
                float* __restrict__ Ypart) {
    const int lane  = threadIdx.x & 31;
    const int wid   = blockIdx.x * 8 + (threadIdx.x >> 5);   // 0..1023
    const int strip = wid >> 2;                              // 0..255
    const int jchunk = wid & (JSPLIT - 1);
    const int g = lane >> 4, r = lane & 15;
    const int i0 = strip * 16;
    const int jbase = jchunk * (N_NODES / JSPLIT);

    v8f acc[5];
#pragma unroll
    for (int t = 0; t < 5; ++t)
        acc[t] = (v8f){0.f, 0.f, 0.f, 0.f, 0.f, 0.f, 0.f, 0.f};

    // ISA A-operand layout (16-bit 16x32): lane(g,r) = row r,
    // halfs 0..7 = K g*8+[0..7], halfs 8..15 = K 16+g*8+[0..7]
    const float* arow = A + (size_t)(i0 + r) * N_NODES + g * 8;

    for (int jt = 0; jt < (N_NODES / JSPLIT) / 32; ++jt) {
        const int j0 = jbase + jt * 32;
        __builtin_prefetch(arow + j0 + 256, 0, 0);   // stream next A lines

        const float4* ap = (const float4*)(arow + j0);
        float4 q0 = ap[0], q1 = ap[1];     // K g*8 + 0..7
        float4 q2 = ap[4], q3 = ap[5];     // K 16 + g*8 + 0..7
        AFrag af;
        af.h[0]  = (_Float16)q0.x; af.h[1]  = (_Float16)q0.y;
        af.h[2]  = (_Float16)q0.z; af.h[3]  = (_Float16)q0.w;
        af.h[4]  = (_Float16)q1.x; af.h[5]  = (_Float16)q1.y;
        af.h[6]  = (_Float16)q1.z; af.h[7]  = (_Float16)q1.w;
        af.h[8]  = (_Float16)q2.x; af.h[9]  = (_Float16)q2.y;
        af.h[10] = (_Float16)q2.z; af.h[11] = (_Float16)q2.w;
        af.h[12] = (_Float16)q3.x; af.h[13] = (_Float16)q3.y;
        af.h[14] = (_Float16)q3.z; af.h[15] = (_Float16)q3.w;

#pragma unroll
        for (int t = 0; t < 5; ++t) {
            // ISA B-operand layout (16-bit 32x16): lane(g,r) = col r,
            // halfs 0..7 = K g*8 + [0..7], halfs 8..15 = K 16 + g*8 + [0..7]
            const uint4* bp = (const uint4*)(Mt + (size_t)(t * 16 + r) * N_NODES
                                                + j0 + g * 8);
            BFrag bf; bf.q[0] = bp[0]; bf.q[1] = bp[2];
            acc[t] = __builtin_amdgcn_wmma_f32_16x16x32_f16(
                false, af.v, false, bf.v, (short)0, acc[t], false, false);
        }
    }

    // D layout: VGPR v, lane(g,r) -> row i0 + v + 8g, col c0 + r
    float* yp = Ypart + (size_t)jchunk * N_NODES * KCOLS;
#pragma unroll
    for (int t = 0; t < 5; ++t)
#pragma unroll
        for (int v = 0; v < 8; ++v)
            yp[(size_t)(i0 + v + 8 * g) * KCOLS + t * 16 + r] = acc[t][v];
}

// ---------------------------------------------------------------------------
// Stage 3: reduce partials, scale by exp(e_l)/denom, ELU, write [4096, 64].
//   denom = (N - deg) + exp(e_l)*S
// ---------------------------------------------------------------------------
__global__ __launch_bounds__(256)
void gat_stage3(const float* __restrict__ Ypart, const float* __restrict__ el_ws,
                float* __restrict__ out) {
    const int idx = blockIdx.x * 256 + threadIdx.x;   // 4096*64
    const int i = idx >> 6, c = idx & 63, h = c >> 3;

    float y = 0.f, S = 0.f, deg = 0.f;
#pragma unroll
    for (int p = 0; p < JSPLIT; ++p) {
        const float* yp = Ypart + ((size_t)p * N_NODES + i) * KCOLS;
        y   += yp[c];
        S   += yp[64 + h];
        deg += yp[72];
    }
    float e     = expf(el_ws[(size_t)h * N_NODES + i]);
    float denom = ((float)N_NODES - deg) + e * S;
    float v     = (e / denom) * y;
    out[idx] = v > 0.f ? v : expm1f(v);
}

// ---------------------------------------------------------------------------
extern "C" void kernel_launch(void* const* d_in, const int* in_sizes, int n_in,
                              void* d_out, int out_size, void* d_ws, size_t ws_size,
                              hipStream_t stream) {
    const float* A  = (const float*)d_in[0];
    const float* H  = (const float*)d_in[1];
    const float* W  = (const float*)d_in[2];
    const float* aL = (const float*)d_in[3];
    const float* aR = (const float*)d_in[4];

    char* ws = (char*)d_ws;
    _Float16* Mt    = (_Float16*)(ws);                         // 80*4096*2   = 655360 B
    float*    el    = (float*)   (ws + 655360);                // 8*4096*4    = 131072 B
    float*    Ypart = (float*)   (ws + 786432);                // 4*4096*80*4 = 5242880 B
    float*    out   = (float*)d_out;

    gat_stage1<<<N_NODES / 4, 256, 0, stream>>>(H, W, aL, aR, Mt, el);
    gat_stage2<<<(N_NODES / 16) * JSPLIT / 8, 256, 0, stream>>>(A, Mt, Ypart);
    gat_stage3<<<N_NODES * 64 / 256, 256, 0, stream>>>(Ypart, el, out);
}
